// MultiHeadedAttention_77249281786308
// MI455X (gfx1250) — matrix operand order repaired and checked by reading, not compiled, not run on hardware
//
#include <hip/hip_runtime.h>
#include <hip/hip_bf16.h>
#include <math.h>

// Problem constants (match reference)
constexpr int CB = 4, CT = 1024, CD = 1024, CH = 16, CDK = 64;
constexpr int CM = CB * CT; // 4096 rows for all [B*T, D] GEMMs

typedef __attribute__((ext_vector_type(16))) __bf16 v16bf;
typedef __attribute__((ext_vector_type(8)))  __bf16 v8bf;
typedef __attribute__((ext_vector_type(8)))  float  v8f;

// ---- element->K order this file uses for WMMA bf16 16x16x32 (contiguous, not the ISA table order) ----
// A: lane half kh=lane>>4: k = 16*kh + e
//    => per-lane contiguous run [16*kh, 16*kh+16)
// B: k = 16*kh + e => per-lane contiguous run of 16
__device__ __forceinline__ void inv_ka(int k, int& kh, int& e) {
    kh = (k >> 4) & 1;
    e  = k & 15;
}
__device__ __forceinline__ v16bf cat8(v8bf a, v8bf b) {
    return __builtin_shufflevector(a, b, 0,1,2,3,4,5,6,7,8,9,10,11,12,13,14,15);
}
__device__ __forceinline__ v16bf ld32B(const __bf16* p) { // 2x global/ds b128
    return cat8(*(const v8bf*)p, *(const v8bf*)(p + 8));
}

// Transpose + convert: Wt[n][k] = (bf16)W[k][n], 1024x1024
__global__ void __launch_bounds__(256)
transpose_bf16_kernel(const float* __restrict__ W, __bf16* __restrict__ Wt) {
    __shared__ float tile[32][33];
    const int n0 = blockIdx.x * 32, k0 = blockIdx.y * 32;
    const int tx = threadIdx.x & 31, ty = threadIdx.x >> 5;
    #pragma unroll
    for (int i = 0; i < 4; ++i)
        tile[ty + i * 8][tx] = W[(size_t)(k0 + ty + i * 8) * CD + n0 + tx];
    __syncthreads();
    #pragma unroll
    for (int i = 0; i < 4; ++i)
        Wt[(size_t)(n0 + ty + i * 8) * CD + k0 + tx] = (__bf16)tile[tx][ty + i * 8];
}

// out = A[M,K] @ Wt^T + bias  (Wt is [N,K] bf16, K innermost)
// mode 0: bf16 split-head [B,H,T,DK]; mode 1: fp32 flat [M,N]; mode 2: bf16 [B,H,DK,T]
__global__ void __launch_bounds__(128)
gemm_bias_kernel(const float* __restrict__ A, const __bf16* __restrict__ Wt,
                 const float* __restrict__ bias, void* __restrict__ outp,
                 int M, int N, int K, int mode) {
    __shared__ alignas(32) __bf16 a_swz[32 * 16]; // A tile in fragment order

    const int tid  = threadIdx.x;
    const int lane = tid & 31;
    const int w    = tid >> 5;
    const int m0   = blockIdx.y * 16;
    const int n0   = blockIdx.x * 64 + w * 16;
    const int mn   = lane & 15;
    const int kh2  = lane >> 4;

    const int ar  = tid >> 3;       // A row 0..15
    const int ak0 = (tid & 7) * 4;  // A col quad base (stays inside one 8-run)

    // B fragment: 16 contiguous bf16 at Wt[n0+mn][k0 + 16*kh2]
    const __bf16* bp = Wt + (size_t)(n0 + mn) * K + (kh2 << 4);

    v8f c = {};
    for (int k0 = 0; k0 < K; k0 += 32) {
        // Stage A tile: one float4 per thread; 4 consecutive e -> merged ds_store_b64
        {
            const float4 f = *(const float4*)&A[(size_t)(m0 + ar) * K + (k0 + ak0)];
            const float fv[4] = {f.x, f.y, f.z, f.w};
            int kh, e0;
            inv_ka(ak0, kh, e0);
            #pragma unroll
            for (int j = 0; j < 4; ++j)
                a_swz[(((kh << 4) | ar) << 4) + e0 + j] = (__bf16)fv[j];
        }
        if (k0 + 32 < K) {
            __builtin_prefetch(&A[(size_t)(m0 + ar) * K + (k0 + 32 + ak0)], 0, 0);
            __builtin_prefetch(bp + k0 + 32, 0, 0);
        }
        __syncthreads();
        const v16bf af = *(const v16bf*)&a_swz[lane << 4]; // 2x ds_load_b128
        const v16bf bf = ld32B(bp + k0);                   // 2x global_load_b128
        c = __builtin_amdgcn_wmma_f32_16x16x32_bf16(false, af, false, bf,
                                                    (short)0, c, false, false);
        __syncthreads();
    }

    const int rowhi = (lane >> 4) * 8;
    #pragma unroll
    for (int r = 0; r < 8; ++r) {
        int gm = m0 + rowhi + r;
        int gn = n0 + mn;
        float val = c[r] + bias[gn];
        if (mode == 1) {
            ((float*)outp)[(size_t)gm * N + gn] = val;
        } else {
            int bI = gm >> 10, tI = gm & (CT - 1);
            int hI = gn >> 6,  dk = gn & (CDK - 1);
            if (mode == 0) // [B,H,T,DK] bf16
                ((__bf16*)outp)[(((size_t)bI * CH + hI) * CT + tI) * CDK + dk] = (__bf16)val;
            else           // mode 2: [B,H,DK,T] bf16 (V transposed for PV B-fragments)
                ((__bf16*)outp)[(((size_t)bI * CH + hI) * CDK + dk) * CT + tI] = (__bf16)val;
        }
    }
}

// One block per (b, h, 16 query rows). 8 waves; wave w owns key-cols [w*128, w*128+128).
__global__ void __launch_bounds__(256)
attn_kernel(const __bf16* __restrict__ Qh, const __bf16* __restrict__ Kh,
            const __bf16* __restrict__ Vt, const int* __restrict__ mask,
            float* __restrict__ p_out, float* __restrict__ xout) {
    __shared__ alignas(32) __bf16 pb_swz[32 * 32 * 16]; // 32 KB, [chunk][lane][e]
    __shared__ float  red[8][16];
    __shared__ float  rowmax[16];
    __shared__ float  rowinv[16];
    __shared__ float  xpart[2][16][64];                 // 8 KB

    const int tid  = threadIdx.x;
    const int lane = tid & 31;
    const int w    = tid >> 5;
    const int b    = blockIdx.z, h = blockIdx.y;
    const int q0   = blockIdx.x * 16;
    const size_t hb = ((size_t)b * CH + h) * (size_t)CT * CDK; // per-head base (same for Vt)

    const int mn    = lane & 15;
    const int kh2   = lane >> 4;
    const int rowhi = kh2 * 8;

    // ---- Q fragments straight from global bf16 (per-lane contiguous 32B runs) ----
    const __bf16* qp = Qh + hb + (size_t)(q0 + mn) * CDK;
    const int kh8 = kh2 * 8;
    const v16bf aq0 = ld32B(qp + (kh2 << 4));
    const v16bf aq1 = ld32B(qp + 32 + (kh2 << 4));

    // ---- Phase 1: scores for this wave's 128 columns, register-resident ----
    float sfrag[8][8];
    unsigned long long mbits = 0ull;
    #pragma unroll
    for (int t = 0; t < 8; ++t) {
        const int kcol0 = (w * 8 + t) * 16;
        const __bf16* kr = Kh + hb + (size_t)(kcol0 + mn) * CDK + (kh2 << 4);
        const v16bf bk0 = ld32B(kr);        // dk 16*kh2..+16 of [0,32)
        const v16bf bk1 = ld32B(kr + 32);   // same of [32,64)
        v8f c = {};
        c = __builtin_amdgcn_wmma_f32_16x16x32_bf16(false, aq0, false, bk0,
                                                    (short)0, c, false, false);
        c = __builtin_amdgcn_wmma_f32_16x16x32_bf16(false, aq1, false, bk1,
                                                    (short)0, c, false, false);
        #pragma unroll
        for (int r = 0; r < 8; ++r) {
            int grow = q0 + rowhi + r;
            int gcol = kcol0 + mn;
            int mk = mask[((size_t)b * CT + grow) * CT + gcol];
            float s = expf(c[r]) * 0.125f; // exp(qk) / sqrt(64)
            if (mk == 0) s = -1e9f; else mbits |= 1ull << (t * 8 + r);
            sfrag[t][r] = s;
        }
    }

    // ---- Row max: shfl within 16-lane groups + cross-wave via LDS ----
    float lred[8];
    #pragma unroll
    for (int r = 0; r < 8; ++r) {
        float v = sfrag[0][r];
        #pragma unroll
        for (int t = 1; t < 8; ++t) v = fmaxf(v, sfrag[t][r]);
        #pragma unroll
        for (int s = 8; s >= 1; s >>= 1) v = fmaxf(v, __shfl_xor(v, s, 16));
        lred[r] = v;
    }
    if (mn == 0) {
        #pragma unroll
        for (int r = 0; r < 8; ++r) red[w][rowhi + r] = lred[r];
    }
    __syncthreads();
    if (tid < 16) {
        float v = red[0][tid];
        #pragma unroll
        for (int i = 1; i < 8; ++i) v = fmaxf(v, red[i][tid]);
        rowmax[tid] = v;
    }
    __syncthreads();

    // ---- Row sum of exp(s - max) ----
    float mymax[8];
    #pragma unroll
    for (int r = 0; r < 8; ++r) mymax[r] = rowmax[rowhi + r];
    #pragma unroll
    for (int r = 0; r < 8; ++r) {
        float v = 0.f;
        #pragma unroll
        for (int t = 0; t < 8; ++t) {
            float e = expf(sfrag[t][r] - mymax[r]); // masked (-1e9) underflows to 0
            sfrag[t][r] = e;
            v += e;
        }
        #pragma unroll
        for (int s = 8; s >= 1; s >>= 1) v += __shfl_xor(v, s, 16);
        lred[r] = v;
    }
    __syncthreads();
    if (mn == 0) {
        #pragma unroll
        for (int r = 0; r < 8; ++r) red[w][rowhi + r] = lred[r];
    }
    __syncthreads();
    if (tid < 16) {
        float v = 0.f;
        #pragma unroll
        for (int i = 0; i < 8; ++i) v += red[i][tid];
        rowinv[tid] = (v > 0.f) ? (1.f / v) : 0.f;
    }
    __syncthreads();

    // ---- Phase 2: p = softmax, re-masked; stream to p_attn, stash swizzled bf16 ----
    #pragma unroll
    for (int t = 0; t < 8; ++t) {
        const int kcol0 = (w * 8 + t) * 16;
        const int col = kcol0 + mn;
        const int cch = col >> 5;
        int kh, e;
        inv_ka(col & 31, kh, e);
        #pragma unroll
        for (int r = 0; r < 8; ++r) {
            int row = rowhi + r;
            float p = sfrag[t][r] * rowinv[row];
            if (!((mbits >> (t * 8 + r)) & 1ull)) p = 0.f;
            p_out[(((size_t)b * CH + h) * CT + (q0 + row)) * CT + col] = p;
            pb_swz[((((cch << 5) | ((kh << 4) | row))) << 4) + e] = (__bf16)p;
        }
    }
    __syncthreads();

    // ---- Phase 3: x = P @ V. Waves 0-3: nt 0-3 over K 0..511; waves 4-7: K 512..1023 ----
    const int nt    = w & 3;
    const int cbase = (w >> 2) * 16;
    // V fragment: Vt is [DK][T] per head -> 16 contiguous bf16 along key index
    const __bf16* vp = Vt + hb + (size_t)(nt * 16 + mn) * CT + (kh2 << 4);
    v8f cx = {};
    for (int ks = 0; ks < 16; ++ks) {
        const int cch = cbase + ks;
        const v16bf ap = *(const v16bf*)&pb_swz[(((cch << 5) | lane)) << 4]; // 2x ds_load_b128
        const v16bf bv = ld32B(vp + cch * 32);                              // 2x global b128
        cx = __builtin_amdgcn_wmma_f32_16x16x32_bf16(false, ap, false, bv,
                                                     (short)0, cx, false, false);
    }
    #pragma unroll
    for (int r = 0; r < 8; ++r)
        xpart[w >> 2][rowhi + r][nt * 16 + mn] = cx[r];
    __syncthreads();

    // Sum K-halves, write context in [B,T,D] layout for the output projection
    for (int i = tid; i < 16 * 64; i += 256) {
        int r = i >> 6, dk = i & 63;
        float v = xpart[0][r][dk] + xpart[1][r][dk];
        xout[((size_t)b * CT + (q0 + r)) * CD + h * CDK + dk] = v;
    }
}

extern "C" void kernel_launch(void* const* d_in, const int* in_sizes, int n_in,
                              void* d_out, int out_size, void* d_ws, size_t ws_size,
                              hipStream_t stream) {
    const float* query = (const float*)d_in[0];
    const float* key   = (const float*)d_in[1];
    const float* value = (const float*)d_in[2];
    const int*   mask  = (const int*)  d_in[3];
    const float* Wq = (const float*)d_in[4];
    const float* bq = (const float*)d_in[5];
    const float* Wk = (const float*)d_in[6];
    const float* bk = (const float*)d_in[7];
    const float* Wv = (const float*)d_in[8];
    const float* bv = (const float*)d_in[9];
    const float* Wo = (const float*)d_in[10];
    const float* bo = (const float*)d_in[11];

    const size_t nElem = (size_t)CM * CD; // 4,194,304
    const size_t wElem = (size_t)CD * CD; // 1,048,576

    // Workspace carve (~48 MB)
    __bf16* Qh  = (__bf16*)d_ws;          // [B,H,T,DK] bf16
    __bf16* Kh  = Qh + nElem;             // [B,H,T,DK] bf16
    __bf16* Vt  = Kh + nElem;             // [B,H,DK,T] bf16
    float*  Xc  = (float*)(Vt + nElem);   // [B,T,D] fp32 context
    __bf16* WqT = (__bf16*)(Xc + nElem);  // [N,K] bf16 transposed weights
    __bf16* WkT = WqT + wElem;
    __bf16* WvT = WkT + wElem;
    __bf16* WoT = WvT + wElem;

    float* out   = (float*)d_out;         // [B,T,D]
    float* p_out = out + nElem;           // [B,H,T,T]

    dim3 tGrid(CD / 32, CD / 32);
    transpose_bf16_kernel<<<tGrid, 256, 0, stream>>>(Wq, WqT);
    transpose_bf16_kernel<<<tGrid, 256, 0, stream>>>(Wk, WkT);
    transpose_bf16_kernel<<<tGrid, 256, 0, stream>>>(Wv, WvT);
    transpose_bf16_kernel<<<tGrid, 256, 0, stream>>>(Wo, WoT);

    dim3 gBlk(128);
    dim3 gGrid(CD / 64, CM / 16); // 16 x 256 blocks
    gemm_bias_kernel<<<gGrid, gBlk, 0, stream>>>(query, WqT, bq, Qh, CM, CD, CD, 0);
    gemm_bias_kernel<<<gGrid, gBlk, 0, stream>>>(key,   WkT, bk, Kh, CM, CD, CD, 0);
    gemm_bias_kernel<<<gGrid, gBlk, 0, stream>>>(value, WvT, bv, Vt, CM, CD, CD, 2);

    dim3 aGrid(CT / 16, CH, CB); // 64 x 16 x 4 blocks
    attn_kernel<<<aGrid, dim3(256), 0, stream>>>(Qh, Kh, Vt, mask, p_out, Xc);

    gemm_bias_kernel<<<gGrid, gBlk, 0, stream>>>(Xc, WoT, bo, out, CM, CD, CD, 1);
}
